// GraphEncoder_44152263803372
// MI455X (gfx1250) — matrix operand order repaired and checked by reading, not compiled, not run on hardware
//
#include <hip/hip_runtime.h>

#define NEG_SLOPE 0.2f

typedef __attribute__((ext_vector_type(16))) _Float16 v16h;
typedef __attribute__((ext_vector_type(8)))  float    v8f;

// ---------------- utility ----------------

__device__ __forceinline__ float atomicMaxFloat(float* addr, float value) {
  if (value >= 0.0f)
    return __int_as_float(atomicMax((int*)addr, __float_as_int(value)));
  else
    return __uint_as_float(atomicMin((unsigned int*)addr, __float_as_uint(value)));
}

__global__ void k_zero_f32(float* __restrict__ p, int n) {
  int i = blockIdx.x * blockDim.x + threadIdx.x;
  if (i < n) p[i] = 0.0f;
}

__global__ void k_fill_f32(float* __restrict__ p, float v, int n) {
  int i = blockIdx.x * blockDim.x + threadIdx.x;
  if (i < n) p[i] = v;
}

__global__ void k_cvt_f16(const float* __restrict__ x, _Float16* __restrict__ y, int n) {
  int i = blockIdx.x * blockDim.x + threadIdx.x;
  if (i < n) y[i] = (_Float16)x[i];
}

// ---------------- self-loop edge_attr = mean of incoming ----------------

__global__ void k_deg_easum(const int* __restrict__ dsts, const float* __restrict__ eattr,
                            float* __restrict__ easum, float* __restrict__ deg, int E) {
  int idx = blockIdx.x * blockDim.x + threadIdx.x;
  if (idx >= E * 16) return;
  int e = idx >> 4, k = idx & 15;
  int d = dsts[e];
  atomicAdd(&easum[d * 16 + k], eattr[idx]);
  if (k == 0) atomicAdd(&deg[d], 1.0f);
}

__global__ void k_loop_attr(float* __restrict__ la, const float* __restrict__ deg, int N) {
  int idx = blockIdx.x * blockDim.x + threadIdx.x;
  if (idx >= N * 16) return;
  la[idx] = la[idx] / fmaxf(deg[idx >> 4], 1.0f);
}

// ---------------- weight pack into WMMA B-operand register layout ----------------
// B (32x16 f16, wave32): half j of lane L holds B[k0 + j + 16*(L>>4)][n0 + (L&15)].
// Packed as P[((tk*8 + tn)*32 + lane)*16 + j] so each lane loads 32 contiguous bytes.

__global__ void k_pack_w(const float* __restrict__ W, _Float16* __restrict__ P) {
  int idx = blockIdx.x * blockDim.x + threadIdx.x;
  if (idx >= 128 * 128) return;
  int j    = idx & 15;
  int lane = (idx >> 4) & 31;
  int tn   = (idx >> 9) & 7;
  int tk   = idx >> 12;
  int k = tk * 32 + j + 16 * (lane >> 4);
  int n = tn * 16 + (lane & 15);
  P[idx] = (_Float16)W[k * 128 + n];
}

// ---------------- fused dual WMMA GEMM: Cl = A@Wl, Cr = A@Wr ----------------
// Block = 256 threads = 8 waves; blockIdx.x = 16-row stripe; wave w owns column tile w.
// A (16x32 f16) layout: half j of lane L = A[m0 + (L&15)][k0 + j + 16*(L>>4)]
//   -> two contiguous 16B loads per lane per K-step; A registers reused for BOTH weight
//   matrices (halves A traffic vs two separate GEMM launches). 8 v_wmma per wave.

__global__ void __launch_bounds__(256)
k_gemm_wmma_dual(const _Float16* __restrict__ A16,
                 const _Float16* __restrict__ Bl, const _Float16* __restrict__ Br,
                 float* __restrict__ Cl, float* __restrict__ Cr, int Nrows) {
  int wave = threadIdx.x >> 5;
  int lane = threadIdx.x & 31;
  int m0 = blockIdx.x * 16;
  if (m0 >= Nrows) return;
  int l15 = lane & 15;
  int hi  = lane >> 4;
  int arow = m0 + l15;
  if (arow >= Nrows) arow = Nrows - 1;

  v8f accl = {}, accr = {};
#pragma unroll
  for (int kt = 0; kt < 4; ++kt) {
    int k0 = kt * 32;
    v16h a, bl, br;
    const _Float16* pa = A16 + (size_t)arow * 128 + k0 + 16 * hi;
    *(uint4*)&a        = *(const uint4*)(pa);        // K offsets 0..7
    *((uint4*)&a + 1)  = *(const uint4*)(pa + 8);    // K offsets 8..15
    size_t boff = ((size_t)(kt * 8 + wave) * 32 + lane) * 16;
    *(uint4*)&bl       = *(const uint4*)(Bl + boff);
    *((uint4*)&bl + 1) = *(const uint4*)(Bl + boff + 8);
    *(uint4*)&br       = *(const uint4*)(Br + boff);
    *((uint4*)&br + 1) = *(const uint4*)(Br + boff + 8);
    accl = __builtin_amdgcn_wmma_f32_16x16x32_f16(false, a, false, bl,
                                                  (short)0, accl, false, false);
    accr = __builtin_amdgcn_wmma_f32_16x16x32_f16(false, a, false, br,
                                                  (short)0, accr, false, false);
  }
  int n0 = wave * 16;
  if (m0 + 16 <= Nrows) {           // full tile: branch-free store path
#pragma unroll
    for (int r = 0; r < 8; ++r) {
      size_t o = (size_t)(m0 + r + 8 * hi) * 128 + n0 + l15;
      Cl[o] = accl[r];
      Cr[o] = accr[r];
    }
  } else {                          // tail stripe only
#pragma unroll
    for (int r = 0; r < 8; ++r) {
      int m = m0 + r + 8 * hi;
      if (m < Nrows) {
        size_t o = (size_t)m * 128 + n0 + l15;
        Cl[o] = accl[r];
        Cr[o] = accr[r];
      }
    }
  }
}

// ---------------- edge kernels (wave32 per edge) ----------------

__global__ void __launch_bounds__(256)
k_edge_logits(const float* __restrict__ xl, const float* __restrict__ xr,
              const int* __restrict__ srcs, const int* __restrict__ dsts,
              const float* __restrict__ eattr, const float* __restrict__ lattr,
              const float* __restrict__ We, const float* __restrict__ att,
              float* __restrict__ sbuf, float* __restrict__ smax, int E, int N) {
  __shared__ float sWe[16 * 128];
  __shared__ float sAtt[128];
  for (int i = threadIdx.x; i < 16 * 128; i += blockDim.x) sWe[i] = We[i];
  for (int i = threadIdx.x; i < 128; i += blockDim.x) sAtt[i] = att[i];
  __syncthreads();

  int lane = threadIdx.x & 31;
  int e = blockIdx.x * 8 + (threadIdx.x >> 5);
  int total = E + N;
  if (e >= total) return;

  int s, d;
  const float* eap;
  if (e < E) { s = srcs[e]; d = dsts[e]; eap = eattr + (size_t)e * 16; }
  else       { s = d = e - E;            eap = lattr + (size_t)(e - E) * 16; }

  float eav = (lane < 16) ? eap[lane] : 0.0f;
  float ek[16];
#pragma unroll
  for (int k = 0; k < 16; ++k) ek[k] = __shfl(eav, k, 32);

  const float* pl = xl + (size_t)s * 128;
  const float* pr = xr + (size_t)d * 128;
  float partial = 0.0f;
#pragma unroll
  for (int u = 0; u < 4; ++u) {
    int f = lane + 32 * u;
    float m = pl[f] + pr[f];
#pragma unroll
    for (int k = 0; k < 16; ++k) m = fmaf(ek[k], sWe[k * 128 + f], m);
    m = (m > 0.0f) ? m : NEG_SLOPE * m;   // leaky relu
    partial = fmaf(sAtt[f], m, partial);
  }
#pragma unroll
  for (int off = 16; off > 0; off >>= 1) partial += __shfl_xor(partial, off, 32);
  if (lane == 0) {
    sbuf[e] = partial;
    atomicMaxFloat(&smax[d], partial);
  }
}

__global__ void k_edge_exp(const int* __restrict__ dsts, float* __restrict__ sbuf,
                           const float* __restrict__ smax, float* __restrict__ denom,
                           int E, int N) {
  int e = blockIdx.x * blockDim.x + threadIdx.x;
  int total = E + N;
  if (e >= total) return;
  int d = (e < E) ? dsts[e] : (e - E);
  float v = expf(sbuf[e] - smax[d]);
  sbuf[e] = v;
  atomicAdd(&denom[d], v);
}

__global__ void __launch_bounds__(256)
k_edge_scatter(const float* __restrict__ xl, const int* __restrict__ srcs,
               const int* __restrict__ dsts, const float* __restrict__ sbuf,
               const float* __restrict__ denom, float* __restrict__ acc, int E, int N) {
  int lane = threadIdx.x & 31;
  int e = blockIdx.x * 8 + (threadIdx.x >> 5);
  int total = E + N;
  if (e >= total) return;
  int s, d;
  if (e < E) { s = srcs[e]; d = dsts[e]; } else { s = d = e - E; }
  float alpha = sbuf[e] / (denom[d] + 1e-16f);
  const float* pl = xl + (size_t)s * 128;
  float* pd = acc + (size_t)d * 128;
#pragma unroll
  for (int u = 0; u < 4; ++u) {
    int f = lane + 32 * u;
    atomicAdd(&pd[f], alpha * pl[f]);
  }
}

__global__ void k_bias_relu(float* __restrict__ acc, const float* __restrict__ b,
                            _Float16* __restrict__ x16, int n) {
  int i = blockIdx.x * blockDim.x + threadIdx.x;
  if (i >= n) return;
  float v = fmaxf(acc[i] + b[i & 127], 0.0f);
  acc[i] = v;               // f32 for pooling (final layer)
  x16[i] = (_Float16)v;     // f16 for next layer's WMMA GEMM
}

// ---------------- pooling ----------------

__global__ void __launch_bounds__(256)
k_pool_acc(const float* __restrict__ x, const int* __restrict__ batch,
           float* __restrict__ out, float* __restrict__ cnt, int N) {
  int lane = threadIdx.x & 31;
  int n = blockIdx.x * 8 + (threadIdx.x >> 5);
  if (n >= N) return;
  int g = batch[n];
#pragma unroll
  for (int u = 0; u < 4; ++u) {
    int f = lane + 32 * u;
    atomicAdd(&out[g * 128 + f], x[(size_t)n * 128 + f]);
  }
  if (lane == 0) atomicAdd(&cnt[g], 1.0f);
}

__global__ void k_pool_div(float* __restrict__ out, const float* __restrict__ cnt, int n) {
  int i = blockIdx.x * blockDim.x + threadIdx.x;
  if (i < n) out[i] = out[i] / fmaxf(cnt[i >> 7], 1.0f);
}

// ---------------- host ----------------

static inline int cdiv(int a, int b) { return (a + b - 1) / b; }

static void run_layer(const _Float16* x16_in, _Float16* x16_scratch,
                      const float* Wl, const float* Wr, const float* We,
                      const float* att, const float* b,
                      _Float16* wlp, _Float16* wrp,
                      float* xl, float* xr, float* acc,
                      float* sbuf, float* smax, float* denom,
                      const int* srcs, const int* dsts,
                      const float* eattr, const float* lattr,
                      int N, int E, hipStream_t stream) {
  int total = E + N;
  k_pack_w<<<cdiv(128 * 128, 256), 256, 0, stream>>>(Wl, wlp);
  k_pack_w<<<cdiv(128 * 128, 256), 256, 0, stream>>>(Wr, wrp);
  k_gemm_wmma_dual<<<cdiv(N, 16), 256, 0, stream>>>(x16_in, wlp, wrp, xl, xr, N);

  k_fill_f32<<<cdiv(N, 256), 256, 0, stream>>>(smax, -__builtin_inff(), N);
  k_zero_f32<<<cdiv(N, 256), 256, 0, stream>>>(denom, N);
  k_zero_f32<<<cdiv(N * 128, 256), 256, 0, stream>>>(acc, N * 128);

  k_edge_logits<<<cdiv(total, 8), 256, 0, stream>>>(xl, xr, srcs, dsts, eattr, lattr,
                                                    We, att, sbuf, smax, E, N);
  k_edge_exp<<<cdiv(total, 256), 256, 0, stream>>>(dsts, sbuf, smax, denom, E, N);
  k_edge_scatter<<<cdiv(total, 8), 256, 0, stream>>>(xl, srcs, dsts, sbuf, denom, acc, E, N);
  k_bias_relu<<<cdiv(N * 128, 256), 256, 0, stream>>>(acc, b, x16_scratch, N * 128);
}

extern "C" void kernel_launch(void* const* d_in, const int* in_sizes, int n_in,
                              void* d_out, int out_size, void* d_ws, size_t ws_size,
                              hipStream_t stream) {
  (void)n_in; (void)out_size; (void)ws_size;

  const float* x0    = (const float*)d_in[0];
  const int*   eidx  = (const int*)d_in[1];
  const int*   batch = (const int*)d_in[2];
  const float* eattr = (const float*)d_in[3];
  const float* Wl1   = (const float*)d_in[4];
  const float* Wr1   = (const float*)d_in[5];
  const float* We1   = (const float*)d_in[6];
  const float* att1  = (const float*)d_in[7];
  const float* b1    = (const float*)d_in[8];
  const float* Wl2   = (const float*)d_in[9];
  const float* Wr2   = (const float*)d_in[10];
  const float* We2   = (const float*)d_in[11];
  const float* att2  = (const float*)d_in[12];
  const float* b2    = (const float*)d_in[13];

  const int N = in_sizes[0] / 128;
  const int E = in_sizes[3] / 16;
  const int total = E + N;
  const int* srcs = eidx;
  const int* dsts = eidx + E;

  // carve workspace
  char* base = (char*)d_ws;
  size_t off = 0;
  auto carve = [&](size_t bytes) -> void* {
    void* p = (void*)(base + off);
    off += (bytes + 255) & ~(size_t)255;
    return p;
  };
  float*    xl    = (float*)carve((size_t)N * 128 * 4);
  float*    xr    = (float*)carve((size_t)N * 128 * 4);
  float*    acc   = (float*)carve((size_t)N * 128 * 4);
  _Float16* x16   = (_Float16*)carve((size_t)N * 128 * 2);
  _Float16* wlp   = (_Float16*)carve(128 * 128 * 2);
  _Float16* wrp   = (_Float16*)carve(128 * 128 * 2);
  float*    sbuf  = (float*)carve((size_t)total * 4);
  float*    smax  = (float*)carve((size_t)N * 4);
  float*    denom = (float*)carve((size_t)N * 4);
  float*    deg   = (float*)carve((size_t)N * 4);
  float*    lattr = (float*)carve((size_t)N * 16 * 4);
  float*    cnt   = (float*)carve(64 * 4);

  // self-loop edge attributes (mean of incoming edge_attr per node)
  k_zero_f32<<<cdiv(N, 256), 256, 0, stream>>>(deg, N);
  k_zero_f32<<<cdiv(N * 16, 256), 256, 0, stream>>>(lattr, N * 16);
  k_deg_easum<<<cdiv(E * 16, 256), 256, 0, stream>>>(dsts, eattr, lattr, deg, E);
  k_loop_attr<<<cdiv(N * 16, 256), 256, 0, stream>>>(lattr, deg, N);

  // layer 1
  k_cvt_f16<<<cdiv(N * 128, 256), 256, 0, stream>>>(x0, x16, N * 128);
  run_layer(x16, x16, Wl1, Wr1, We1, att1, b1, wlp, wrp,
            xl, xr, acc, sbuf, smax, denom, srcs, dsts, eattr, lattr, N, E, stream);
  // layer 2 (x16 now holds relu(layer1 out) in f16)
  run_layer(x16, x16, Wl2, Wr2, We2, att2, b2, wlp, wrp,
            xl, xr, acc, sbuf, smax, denom, srcs, dsts, eattr, lattr, N, E, stream);

  // global mean pool over batch (acc holds f32 relu'd layer-2 output)
  float* out = (float*)d_out;
  k_zero_f32<<<cdiv(64 * 128, 256), 256, 0, stream>>>(out, 64 * 128);
  k_zero_f32<<<1, 64, 0, stream>>>(cnt, 64);
  k_pool_acc<<<cdiv(N, 8), 256, 0, stream>>>(acc, batch, out, cnt, N);
  k_pool_div<<<cdiv(64 * 128, 256), 256, 0, stream>>>(out, cnt, 64 * 128);
}
